// Encoder_16638703305091
// MI455X (gfx1250) — matrix operand order repaired and checked by reading, not compiled, not run on hardware
//
#include <hip/hip_runtime.h>
#include <hip/hip_bf16.h>

typedef __attribute__((ext_vector_type(16))) _Float16 v16h;
typedef __attribute__((ext_vector_type(8)))  float    v8f;

#define D_MODEL  512
#define N_HEADS  8
#define HEAD_DIM 64
#define SEQ      8192
#define BATCH    2
#define D_HIDDEN 2048
#define MTOT     (BATCH * SEQ)      // 16384 rows
#define BH       (BATCH * N_HEADS)  // 16
#define WIN      288                // key window per 16-row strip (multiple of 32)

static __device__ __forceinline__ v8f wmma16(v16h a, v16h b, v8f c) {
    // v_wmma_f32_16x16x32_f16, f32 accumulate
    return __builtin_amdgcn_wmma_f32_16x16x32_f16(false, a, false, b, (short)0, c,
                                                  false, false);
}

// ---------------------------------------------------------------------------
// A-fragment layout (16-bit, 16x32): lane L, elem i -> row = L&15,
//   k = (i<8 ? (L>>4)*8 + i : 16 + (L>>4)*8 + (i-8))
// B-fragment layout (16-bit, 32x16): lane L, elem i -> col = L&15,
//   k = (i>>3)*16 + (L>>4)*8 + (i&7)
// C/D (f32 16x16): lane L, vgpr r -> row = (L>>4)*8 + r, col = L&15
// ---------------------------------------------------------------------------

// f32 row-major [M x K] -> f16 A-fragment tiles ((mt*KT+kt)*32 + lane)*16 + i
__global__ void cast_swizzle_A(const float* __restrict__ src,
                               _Float16* __restrict__ dst, int M, int K) {
    long idx = (long)blockIdx.x * blockDim.x + threadIdx.x;
    long total = (long)M * K;
    if (idx >= total) return;
    int  i    = (int)(idx & 15);
    int  lane = (int)((idx >> 4) & 31);
    long tile = idx >> 9;
    int  KT = K >> 5;
    int  mt = (int)(tile / KT), kt = (int)(tile % KT);
    int  row = mt * 16 + (lane & 15);
    int  b0 = lane >> 4, b1 = i >> 3, j = i & 7;
    int  col = kt * 32 + b1 * 16 + b0 * 8 + j;
    dst[idx] = (_Float16)src[(long)row * K + col];
}

// f32 row-major [K x N] -> f16 B-fragment tiles ((kt*NT+nt)*32 + lane)*16 + i
__global__ void cast_swizzle_B(const float* __restrict__ src,
                               _Float16* __restrict__ dst, int K, int N) {
    long idx = (long)blockIdx.x * blockDim.x + threadIdx.x;
    long total = (long)K * N;
    if (idx >= total) return;
    int  i    = (int)(idx & 15);
    int  lane = (int)((idx >> 4) & 31);
    long tile = idx >> 9;
    int  NT = N >> 4;
    int  kt = (int)(tile / NT), nt = (int)(tile % NT);
    int  col = nt * 16 + (lane & 15);
    int  kk  = kt * 32 + (i >> 3) * 16 + (lane >> 4) * 8 + (i & 7);
    dst[idx] = (_Float16)src[(long)kk * N + col];
}

// ---------------------------------------------------------------------------
// Register-blocked MMA worker: one wave computes a 32x64 output tile
// (2 M-tiles x 4 N-tiles), 8 WMMAs per k-step on 6 fragment loads
// (~21 flop/byte of L2 fragment traffic vs 8 for a single 16x16 tile).
// ---------------------------------------------------------------------------
template <int KT, int NT>
static __device__ __forceinline__ void mma_block(
    const _Float16* __restrict__ afrag, const _Float16* __restrict__ bfrag,
    int mt0, int nt0, int lane, v8f acc[2][4]) {
    const _Float16* ap0 = afrag + (((long)(mt0 + 0) * KT) * 32 + lane) * 16;
    const _Float16* ap1 = afrag + (((long)(mt0 + 1) * KT) * 32 + lane) * 16;
    const _Float16* bp  = bfrag + (((long)nt0) * 32 + lane) * 16;
#pragma unroll 4
    for (int kt = 0; kt < KT; ++kt) {
        v16h a0 = *(const v16h*)(ap0 + (long)kt * 512);
        v16h a1 = *(const v16h*)(ap1 + (long)kt * 512);
        const _Float16* bk = bp + (long)kt * NT * 512;
        v16h b0 = *(const v16h*)(bk);
        v16h b1 = *(const v16h*)(bk + 512);
        v16h b2 = *(const v16h*)(bk + 1024);
        v16h b3 = *(const v16h*)(bk + 1536);
        acc[0][0] = wmma16(a0, b0, acc[0][0]);
        acc[0][1] = wmma16(a0, b1, acc[0][1]);
        acc[0][2] = wmma16(a0, b2, acc[0][2]);
        acc[0][3] = wmma16(a0, b3, acc[0][3]);
        acc[1][0] = wmma16(a1, b0, acc[1][0]);
        acc[1][1] = wmma16(a1, b1, acc[1][1]);
        acc[1][2] = wmma16(a1, b2, acc[1][2]);
        acc[1][3] = wmma16(a1, b3, acc[1][3]);
    }
}

// ---------------------------------------------------------------------------
// QKV projection GEMM: [MTOT x 512] @ [512 x 512] + bias, optional scale,
// output f16 in head-major [BH, SEQ, 64] layout. 8 waves/block, 32x64/wave.
// ---------------------------------------------------------------------------
__global__ __launch_bounds__(256) void qkv_gemm(
    const _Float16* __restrict__ afrag, const _Float16* __restrict__ bfrag,
    const float* __restrict__ bias, _Float16* __restrict__ out, float scale) {
    const int KT = D_MODEL / 32;  // 16
    const int NT = D_MODEL / 16;  // 32
    const int NGRP = NT / 4;      // 8 groups of 4 N-tiles
    int  wave = threadIdx.x >> 5, lane = threadIdx.x & 31;
    long tid = (long)blockIdx.x * 8 + wave;
    int  mt0 = (int)(tid / NGRP) * 2;
    int  nt0 = (int)(tid % NGRP) * 4;

    v8f acc[2][4] = {};
    mma_block<KT, NT>(afrag, bfrag, mt0, nt0, lane, acc);

    int rbase = (lane >> 4) * 8;
#pragma unroll
    for (int j = 0; j < 4; ++j) {
        int   coln = (nt0 + j) * 16 + (lane & 15);
        int   h = coln >> 6, d = coln & 63;
        float bv = bias[coln];
#pragma unroll
        for (int i = 0; i < 2; ++i) {
#pragma unroll
            for (int r = 0; r < 8; ++r) {
                int   grow = (mt0 + i) * 16 + rbase + r;
                int   bb = grow >> 13;          // / SEQ
                int   s  = grow & (SEQ - 1);
                float v  = (acc[i][j] [r] + bv) * scale;
                out[(((long)(bb * N_HEADS + h)) * SEQ + s) * HEAD_DIM + d] =
                    (_Float16)v;
            }
        }
    }
}

// ---------------------------------------------------------------------------
// Sliding-window attention. 2 waves/block; each wave owns one (bh, 16-row
// strip). 288-wide key window = the +/-128 band for all 16 rows.
// ---------------------------------------------------------------------------
__global__ __launch_bounds__(64) void attn_kernel(
    const _Float16* __restrict__ q, const _Float16* __restrict__ k,
    const _Float16* __restrict__ v, float* __restrict__ attn_out,
    float* __restrict__ band_out) {
    __shared__ float    sc[2][16][WIN];   // raw scores -> exp values (f32)
    __shared__ _Float16 pr[2][16][WIN];   // normalized probs (f16)

    int  wave = threadIdx.x >> 5, lane = threadIdx.x & 31;
    long sid = (long)blockIdx.x * 2 + wave;
    int  bh = (int)(sid >> 9);            // / (SEQ/16)
    int  mt = (int)(sid & 511);
    int  s0 = mt * 16;
    int  key0 = s0 - 128;                 // window start (may be negative)

    const _Float16* qb = q + (long)bh * SEQ * HEAD_DIM;
    const _Float16* kb = k + (long)bh * SEQ * HEAD_DIM;
    const _Float16* vb = v + (long)bh * SEQ * HEAD_DIM;

    // --- q A-fragments for the two 32-wide halves of D=64 (contiguous) ---
    int arow  = s0 + (lane & 15);
    int khalf = (lane >> 4) * 16;
    const _Float16* qrow = qb + (long)arow * HEAD_DIM;
    v16h aq0, aq1;
#pragma unroll
    for (int i = 0; i < 8; ++i) {
        aq0[i]     = qrow[khalf + i];
        aq0[8 + i] = qrow[8 + khalf + i];
        aq1[i]     = qrow[32 + khalf + i];
        aq1[8 + i] = qrow[40 + khalf + i];
    }

    // --- scores = q @ k^T over 18 key tiles of 16 ---
    int kb16 = (lane >> 4) * 16;
    for (int ct = 0; ct < 18; ++ct) {
        int keyi = key0 + ct * 16 + (lane & 15);
        int keyc = keyi < 0 ? 0 : (keyi >= SEQ ? SEQ - 1 : keyi);
        const _Float16* krow = kb + (long)keyc * HEAD_DIM;
        v16h bk0 = *(const v16h*)(krow + kb16);        // d = kb16..kb16+15
        v16h bk1 = *(const v16h*)(krow + 32 + kb16);   // d = 32+kb16..
        v8f s = {};
        s = wmma16(aq0, bk0, s);
        s = wmma16(aq1, bk1, s);
        int rbase = (lane >> 4) * 8;
        int ccol  = ct * 16 + (lane & 15);
#pragma unroll
        for (int r = 0; r < 8; ++r) {
            int  rr  = rbase + r;
            int  qg  = s0 + rr;
            int  kg  = key0 + ccol;
            int  dlt = kg - qg;
            bool ok = (dlt >= -128) && (dlt <= 128) && (kg >= 0) && (kg < SEQ);
            sc[wave][rr][ccol] = ok ? s[r] : -1.0e9f;
        }
    }
    __syncthreads();

    // --- f32 softmax, one row per lane (lanes 0..15); write band output ---
    if (lane < 16) {
        int   rr = lane;
        float m = -3.0e38f;
        for (int c = 0; c < WIN; ++c) m = fmaxf(m, sc[wave][rr][c]);
        float sum = 0.f;
        for (int c = 0; c < WIN; ++c) {
            float e = expf(sc[wave][rr][c] - m);
            sc[wave][rr][c] = e;
            sum += e;
        }
        float inv = 1.0f / sum;
        for (int c = 0; c < WIN; ++c)
            pr[wave][rr][c] = (_Float16)(sc[wave][rr][c] * inv);
        int    qg = s0 + rr;
        float* bp = band_out + ((long)bh * SEQ + qg) * 257;
        for (int j = 0; j <= 256; ++j)                  // band col = rr + j
            bp[j] = sc[wave][rr][rr + j] * inv;
    }
    __syncthreads();

    // --- out = probs @ v : N=64 (4 tiles), K=288 (9 k-steps) ---
    int arow2  = lane & 15;
    int khalf2 = (lane >> 4) * 8;
    int bidx = bh >> 3, hh = bh & 7;
    for (int nt = 0; nt < 4; ++nt) {
        v8f acc = {};
        int col = nt * 16 + (lane & 15);
        for (int kt = 0; kt < 9; ++kt) {
            int  k0 = kt * 32;
            v16h a;
#pragma unroll
            for (int i = 0; i < 8; ++i) {
                a[i]     = pr[wave][arow2][k0 + khalf2 + i];
                a[8 + i] = pr[wave][arow2][k0 + 16 + khalf2 + i];
            }
            v16h b;
#pragma unroll
            for (int i = 0; i < 16; ++i) {
                int kg = key0 + k0 + (i >> 3) * 16 + (lane >> 4) * 8 + (i & 7);
                int kc = kg < 0 ? 0 : (kg >= SEQ ? SEQ - 1 : kg);
                b[i] = vb[(long)kc * HEAD_DIM + col];   // prob==0 masks OOB
            }
            acc = wmma16(a, b, acc);
        }
        int rbase = (lane >> 4) * 8;
        int coln  = hh * HEAD_DIM + col;
#pragma unroll
        for (int r = 0; r < 8; ++r) {
            int s = s0 + rbase + r;
            attn_out[((long)bidx * SEQ + s) * D_MODEL + coln] = acc[r];
        }
    }
}

// ---------------------------------------------------------------------------
// Residual add + layernorm over 512; optionally also emit f16 A-fragments.
// One block (256 threads) per row.
// ---------------------------------------------------------------------------
static __device__ __forceinline__ void scatterA(_Float16* frag, int K, int row,
                                                int col, float val) {
    int KT = K >> 5;
    int mt = row >> 4, rr = row & 15;
    int kt = col >> 5, kin = col & 31;
    int lane = rr + 16 * ((kin >> 3) & 1);
    int i    = 8 * ((kin >> 4) & 1) + (kin & 7);
    frag[(((long)mt * KT + kt) * 32 + lane) * 16 + i] = (_Float16)val;
}

__global__ __launch_bounds__(256) void add_ln_kernel(
    const float* __restrict__ a, const float* __restrict__ b,
    const float* __restrict__ g, const float* __restrict__ beta,
    float* __restrict__ out, _Float16* outfrag) {
    __shared__ float rsum[256], rsq[256];
    int  t = threadIdx.x;
    long base = (long)blockIdx.x * D_MODEL;
    float x0 = a[base + t];
    float x1 = a[base + 256 + t];
    if (b) { x0 += b[base + t]; x1 += b[base + 256 + t]; }
    rsum[t] = x0 + x1;
    rsq[t]  = x0 * x0 + x1 * x1;
    __syncthreads();
    for (int off = 128; off > 0; off >>= 1) {
        if (t < off) { rsum[t] += rsum[t + off]; rsq[t] += rsq[t + off]; }
        __syncthreads();
    }
    float mu  = rsum[0] * (1.0f / D_MODEL);
    float var = rsq[0] * (1.0f / D_MODEL) - mu * mu;
    float inv = rsqrtf(var + 1e-5f);
    float y0 = (x0 - mu) * inv * g[t] + beta[t];
    float y1 = (x1 - mu) * inv * g[256 + t] + beta[256 + t];
    out[base + t] = y0;
    out[base + 256 + t] = y1;
    if (outfrag) {
        scatterA(outfrag, D_MODEL, blockIdx.x, t, y0);
        scatterA(outfrag, D_MODEL, blockIdx.x, 256 + t, y1);
    }
}

// ---------------------------------------------------------------------------
// FFN1: [MTOT x 512] @ [512 x 2048], bias + relu, emit f16 A-fragments (K=2048)
// ---------------------------------------------------------------------------
__global__ __launch_bounds__(256) void ffn1_gemm(
    const _Float16* __restrict__ afrag, const _Float16* __restrict__ bfrag,
    const float* __restrict__ bias, _Float16* __restrict__ outfrag) {
    const int KT = D_MODEL / 32;   // 16
    const int NT = D_HIDDEN / 16;  // 128
    const int NGRP = NT / 4;       // 32
    int  wave = threadIdx.x >> 5, lane = threadIdx.x & 31;
    long tid = (long)blockIdx.x * 8 + wave;
    int  mt0 = (int)(tid / NGRP) * 2;
    int  nt0 = (int)(tid % NGRP) * 4;

    v8f acc[2][4] = {};
    mma_block<KT, NT>(afrag, bfrag, mt0, nt0, lane, acc);

    int rbase = (lane >> 4) * 8;
#pragma unroll
    for (int j = 0; j < 4; ++j) {
        int   col = (nt0 + j) * 16 + (lane & 15);
        float bb  = bias[col];
        int   kt2 = col >> 5, kin = col & 31;
        int   b0 = (kin >> 3) & 1, b1 = (kin >> 4) & 1, jj = kin & 7;
        int   i2 = 8 * b1 + jj;
#pragma unroll
        for (int i = 0; i < 2; ++i) {
#pragma unroll
            for (int r = 0; r < 8; ++r) {
                int   rr = rbase + r;
                float vv = fmaxf(acc[i][j] [r] + bb, 0.0f);
                int   lane2 = rr + 16 * b0;
                outfrag[(((long)(mt0 + i) * (D_HIDDEN / 32) + kt2) * 32 + lane2) *
                            16 + i2] = (_Float16)vv;
            }
        }
    }
}

// ---------------------------------------------------------------------------
// FFN2: [MTOT x 2048] @ [2048 x 512] + bias + residual(h) -> f32
// ---------------------------------------------------------------------------
__global__ __launch_bounds__(256) void ffn2_gemm(
    const _Float16* __restrict__ afrag, const _Float16* __restrict__ bfrag,
    const float* __restrict__ bias, const float* __restrict__ hres,
    float* __restrict__ y) {
    const int KT = D_HIDDEN / 32;  // 64
    const int NT = D_MODEL / 16;   // 32
    const int NGRP = NT / 4;       // 8
    int  wave = threadIdx.x >> 5, lane = threadIdx.x & 31;
    long tid = (long)blockIdx.x * 8 + wave;
    int  mt0 = (int)(tid / NGRP) * 2;
    int  nt0 = (int)(tid % NGRP) * 4;

    v8f acc[2][4] = {};
    mma_block<KT, NT>(afrag, bfrag, mt0, nt0, lane, acc);

    int rbase = (lane >> 4) * 8;
#pragma unroll
    for (int j = 0; j < 4; ++j) {
        int   col = (nt0 + j) * 16 + (lane & 15);
        float bb  = bias[col];
#pragma unroll
        for (int i = 0; i < 2; ++i) {
#pragma unroll
            for (int r = 0; r < 8; ++r) {
                long row = (long)(mt0 + i) * 16 + rbase + r;
                y[row * D_MODEL + col] =
                    acc[i][j] [r] + bb + hres[row * D_MODEL + col];
            }
        }
    }
}

// ---------------------------------------------------------------------------
extern "C" void kernel_launch(void* const* d_in, const int* in_sizes, int n_in,
                              void* d_out, int out_size, void* d_ws,
                              size_t ws_size, hipStream_t stream) {
    (void)in_sizes; (void)n_in; (void)out_size; (void)ws_size;
    const float* x    = (const float*)d_in[0];
    const float* Wq   = (const float*)d_in[1];
    const float* bq   = (const float*)d_in[2];
    const float* Wk   = (const float*)d_in[3];
    const float* bk   = (const float*)d_in[4];
    const float* Wv   = (const float*)d_in[5];
    const float* bv   = (const float*)d_in[6];
    const float* ln1g = (const float*)d_in[7];
    const float* ln1b = (const float*)d_in[8];
    const float* Wf1  = (const float*)d_in[9];
    const float* bf1  = (const float*)d_in[10];
    const float* Wf2  = (const float*)d_in[11];
    const float* bf2  = (const float*)d_in[12];
    const float* ln2g = (const float*)d_in[13];
    const float* ln2b = (const float*)d_in[14];

    char*  ws  = (char*)d_ws;
    size_t off = 0;
    auto alloc = [&](size_t bytes) -> char* {
        char* p = ws + off;
        off += (bytes + 255) & ~(size_t)255;
        return p;
    };
    _Float16* xfrag   = (_Float16*)alloc((size_t)MTOT * D_MODEL * 2);   // reused as hfrag
    _Float16* wqf     = (_Float16*)alloc((size_t)D_MODEL * D_MODEL * 2);
    _Float16* wkf     = (_Float16*)alloc((size_t)D_MODEL * D_MODEL * 2);
    _Float16* wvf     = (_Float16*)alloc((size_t)D_MODEL * D_MODEL * 2);
    _Float16* wf1f    = (_Float16*)alloc((size_t)D_MODEL * D_HIDDEN * 2);
    _Float16* wf2f    = (_Float16*)alloc((size_t)D_HIDDEN * D_MODEL * 2);
    _Float16* q16     = (_Float16*)alloc((size_t)BH * SEQ * HEAD_DIM * 2);
    _Float16* k16     = (_Float16*)alloc((size_t)BH * SEQ * HEAD_DIM * 2);
    _Float16* v16     = (_Float16*)alloc((size_t)BH * SEQ * HEAD_DIM * 2);
    float*    attnf   = (float*)alloc((size_t)MTOT * D_MODEL * 4);      // reused as ybuf
    float*    hf32    = (float*)alloc((size_t)MTOT * D_MODEL * 4);
    _Float16* ff1frag = (_Float16*)alloc((size_t)MTOT * D_HIDDEN * 2);
    _Float16* hfrag   = xfrag;   // x fragments dead after QKV GEMMs
    float*    ybuf    = attnf;   // attn dead after LN1

    float* out_main = (float*)d_out;
    float* out_band = out_main + (size_t)MTOT * D_MODEL;  // (B,H,S,257)

    // 1) cast/swizzle activations + weights to WMMA fragment layouts
    cast_swizzle_A<<<(MTOT * D_MODEL + 255) / 256, 256, 0, stream>>>(x, xfrag, MTOT, D_MODEL);
    cast_swizzle_B<<<(D_MODEL * D_MODEL + 255) / 256, 256, 0, stream>>>(Wq, wqf, D_MODEL, D_MODEL);
    cast_swizzle_B<<<(D_MODEL * D_MODEL + 255) / 256, 256, 0, stream>>>(Wk, wkf, D_MODEL, D_MODEL);
    cast_swizzle_B<<<(D_MODEL * D_MODEL + 255) / 256, 256, 0, stream>>>(Wv, wvf, D_MODEL, D_MODEL);
    cast_swizzle_B<<<(D_MODEL * D_HIDDEN + 255) / 256, 256, 0, stream>>>(Wf1, wf1f, D_MODEL, D_HIDDEN);
    cast_swizzle_B<<<(D_HIDDEN * D_MODEL + 255) / 256, 256, 0, stream>>>(Wf2, wf2f, D_HIDDEN, D_MODEL);

    // 2) QKV projections (WMMA), q pre-scaled by 1/sqrt(HEAD_DIM)
    // waves = (MTOT/32) * (D_MODEL/64) = 512*8 = 4096 -> 512 blocks of 8 waves
    qkv_gemm<<<512, 256, 0, stream>>>(xfrag, wqf, bq, q16, 0.125f);
    qkv_gemm<<<512, 256, 0, stream>>>(xfrag, wkf, bk, k16, 1.0f);
    qkv_gemm<<<512, 256, 0, stream>>>(xfrag, wvf, bv, v16, 1.0f);

    // 3) sliding-window attention (WMMA scores + WMMA probs@v), band output
    attn_kernel<<<BH * (SEQ / 16) / 2, 64, 0, stream>>>(q16, k16, v16, attnf, out_band);

    // 4) h = LN(attn + x); also emit h as f16 A-fragments for FFN
    add_ln_kernel<<<MTOT, 256, 0, stream>>>(attnf, x, ln1g, ln1b, hf32, hfrag);

    // 5) FFN
    // FFN1 waves = (MTOT/32)*(D_HIDDEN/64) = 512*32 = 16384 -> 2048 blocks
    ffn1_gemm<<<2048, 256, 0, stream>>>(hfrag, wf1f, bf1, ff1frag);
    // FFN2 waves = (MTOT/32)*(D_MODEL/64) = 512*8 = 4096 -> 512 blocks
    ffn2_gemm<<<512, 256, 0, stream>>>(ff1frag, wf2f, bf2, hf32, ybuf);

    // 6) out = LN(ff + h)
    add_ln_kernel<<<MTOT, 256, 0, stream>>>(ybuf, nullptr, ln2g, ln2b, out_main, nullptr);
}
